// EquivariantConvolution_18167711662487
// MI455X (gfx1250) — matrix operand order repaired and checked by reading, not compiled, not run on hardware
//
#include <hip/hip_runtime.h>

typedef __attribute__((ext_vector_type(16))) _Float16 v16h;
typedef __attribute__((ext_vector_type(8)))  float    v8f;

#define E_TILE   64
#define EDGE_DIM 32
#define HIDDEN   128
#define M1R      16
#define M2R      16
#define D1R      3
#define D2R      3
#define NREPS    3
#define JDIM     48          // M1*NREPS
#define NOUT     768         // M2*M1*NREPS

// LDS row strides (in halves / floats) padded to avoid bank conflicts
#define EF_STRIDE  40        // 32 + 8 halves
#define H_STRIDE   136       // 128 + 8 halves
#define TMP_STRIDE 144       // 48*3 floats

// ---- fragment loaders -------------------------------------------------------
// A (MxK, 16-bit, 16x32): lane q=row; halves [off..off+7] and [off+16..off+23],
// off = kk*32 + hi*8  (ISA 7.12.2 16-bit A layout)
__device__ __forceinline__ v16h load_a_frag(const _Float16* row, int off) {
  v16h r;
  ((uint4*)&r)[0] = *((const uint4*)(row + off));
  ((uint4*)&r)[1] = *((const uint4*)(row + off + 16));
  return r;
}
// B (KxN, 16-bit, 32x16), stored as Bt[n][k]: lane q=col; halves [off..off+7]
// and [off+16..off+23], off = kk*32+hi*8 (same K order as A)
__device__ __forceinline__ v16h load_b_frag(const _Float16* row, int off) {
  v16h r;
  ((uint4*)&r)[0] = *((const uint4*)(row + off));
  ((uint4*)&r)[1] = *((const uint4*)(row + off + 16));
  return r;
}

#define WMMA_F16(A, B, C)                                                      \
  __builtin_amdgcn_wmma_f32_16x16x32_f16(false, (A), false, (B), (short)0,     \
                                         (C), false, false)

// ---- prep: fp32 weights -> f16, transposed to [N][K] ------------------------
__global__ void ec_prep_kernel(const float* __restrict__ W1,
                               const float* __restrict__ W2,
                               _Float16* __restrict__ W1t,
                               _Float16* __restrict__ W2t) {
  int idx = blockIdx.x * blockDim.x + threadIdx.x;
  if (idx < EDGE_DIM * HIDDEN) {            // W1 [32][128] -> W1t [128][32]
    int k = idx / HIDDEN, n = idx % HIDDEN;
    W1t[n * EDGE_DIM + k] = (_Float16)W1[idx];
  }
  if (idx < HIDDEN * NOUT) {                // W2 [128][768] -> W2t [768][128]
    int k = idx / NOUT, n = idx % NOUT;
    W2t[n * HIDDEN + k] = (_Float16)W2[idx];
  }
}

// ---- main fused kernel ------------------------------------------------------
__launch_bounds__(256)
__global__ void ec_main_kernel(const int*   __restrict__ U,
                               const float* __restrict__ basis,
                               const float* __restrict__ edge_feats,
                               const float* __restrict__ f,
                               const float* __restrict__ b1,
                               const float* __restrict__ b2,
                               const _Float16* __restrict__ W1t,
                               const _Float16* __restrict__ W2t,
                               float* __restrict__ out,
                               int E) {
  __shared__ _Float16 sEF[E_TILE * EF_STRIDE];   // edge_feats tile (f16)
  __shared__ _Float16 sH [E_TILE * H_STRIDE];    // hidden activations (f16)
  __shared__ float    sTMP[E_TILE * TMP_STRIDE]; // tmp[e][j][d]
  __shared__ int      sU[E_TILE];

  const int tid = threadIdx.x;
  const int e0  = blockIdx.x * E_TILE;
  const int w   = tid >> 5;       // wave id 0..7
  const int l   = tid & 31;       // lane
  const int q   = l & 15;         // lane%16
  const int hi  = l >> 4;         // lane half

  // ---- cooperative loads ----
  if (tid < E_TILE) {
    int e = e0 + tid;
    sU[tid] = (e < E) ? U[e] : 0;
  }
  for (int idx = tid; idx < E_TILE * EDGE_DIM; idx += 256) {
    int e = idx >> 5, c = idx & 31;
    float v = (e0 + e < E) ? edge_feats[(size_t)(e0 + e) * EDGE_DIM + c] : 0.f;
    sEF[e * EF_STRIDE + c] = (_Float16)v;
  }
  // prefetch next block's edge_feats tile into GL2 while we crunch this one
  {
    int ne = e0 + E_TILE;
    if (ne + (tid >> 2) < E)
      __builtin_prefetch(edge_feats + (size_t)(ne + (tid >> 2)) * EDGE_DIM +
                             (tid & 3) * 8, 0, 1);
  }
  __syncthreads();

  // ---- GEMM1: h = relu(ef @ W1 + b1); [64x32]@[32x128], one WMMA per tile ----
  for (int job = w; job < 4 * (HIDDEN / 16); job += 8) {
    int mt = job >> 3, nt = job & 7;
    v16h a = load_a_frag(sEF + (mt * 16 + q) * EF_STRIDE, hi * 8);
    v16h b = load_b_frag(W1t + (size_t)(nt * 16 + q) * EDGE_DIM, hi * 8);
    v8f acc = {};
    acc = WMMA_F16(a, b, acc);
    int n = nt * 16 + q;
    float bias = b1[n];
#pragma unroll
    for (int v = 0; v < 8; ++v) {
      int el = mt * 16 + v + 8 * hi;
      float x = acc[v] + bias;
      x = x > 0.f ? x : 0.f;
      sH[el * H_STRIDE + n] = (_Float16)x;
    }
  }

  // ---- tmp[e, m1*3+rep, d2] = sum_d1 f[U[e],m1,d1] * basis[e,d1,rep*3+d2] ----
  for (int idx = tid; idx < E_TILE * JDIM * D2R; idx += 256) {
    int e  = idx / (JDIM * D2R);
    int r  = idx - e * (JDIM * D2R);
    int j  = r / D2R, d2 = r - j * D2R;
    int m1 = j / NREPS, rep = j - m1 * NREPS;
    int ge = e0 + e;
    const float* frow = f + ((size_t)sU[e] * M1R + m1) * D1R;
    const float* brow = basis + (size_t)((ge < E) ? ge : 0) * (D1R * NREPS * D2R);
    float s = 0.f;
#pragma unroll
    for (int d1 = 0; d1 < D1R; ++d1)
      s += frow[d1] * brow[d1 * (NREPS * D2R) + rep * D2R + d2];
    sTMP[e * TMP_STRIDE + j * D2R + d2] = s;
  }
  __syncthreads();

  // ---- GEMM2 (rw = h@W2+b2) fused with out = rw @ tmp ----
  // Loop order: i (outer) -> kk -> load 3 B frags once -> all 4 edge subtiles.
  // Each W2t fragment is fetched exactly once per block (192 KB = one W2 pass).
#pragma unroll
  for (int ii = 0; ii < 2; ++ii) {
    const int i = w + 8 * ii;       // output channel 0..15
    v8f c00 = {}, c01 = {}, c02 = {};
    v8f c10 = {}, c11 = {}, c12 = {};
    v8f c20 = {}, c21 = {}, c22 = {};
    v8f c30 = {}, c31 = {}, c32 = {};
    const _Float16* b2row = W2t + (size_t)(i * 48 + q) * HIDDEN;
#pragma unroll
    for (int kk = 0; kk < 4; ++kk) {
      const int aoff = kk * 32 + hi * 8;
      const int boff = kk * 32 + hi * 8;
      v16h bf0 = load_b_frag(b2row,                boff);
      v16h bf1 = load_b_frag(b2row + 16 * HIDDEN,  boff);
      v16h bf2 = load_b_frag(b2row + 32 * HIDDEN,  boff);
      {
        v16h a = load_a_frag(sH + (0 * 16 + q) * H_STRIDE, aoff);
        c00 = WMMA_F16(a, bf0, c00);
        c01 = WMMA_F16(a, bf1, c01);
        c02 = WMMA_F16(a, bf2, c02);
      }
      {
        v16h a = load_a_frag(sH + (1 * 16 + q) * H_STRIDE, aoff);
        c10 = WMMA_F16(a, bf0, c10);
        c11 = WMMA_F16(a, bf1, c11);
        c12 = WMMA_F16(a, bf2, c12);
      }
      {
        v16h a = load_a_frag(sH + (2 * 16 + q) * H_STRIDE, aoff);
        c20 = WMMA_F16(a, bf0, c20);
        c21 = WMMA_F16(a, bf1, c21);
        c22 = WMMA_F16(a, bf2, c22);
      }
      {
        v16h a = load_a_frag(sH + (3 * 16 + q) * H_STRIDE, aoff);
        c30 = WMMA_F16(a, bf0, c30);
        c31 = WMMA_F16(a, bf1, c31);
        c32 = WMMA_F16(a, bf2, c32);
      }
    }

    // epilogue per edge subtile: out[e,i,d] = sum_j (rw[e,i,j]) * tmp[e,j,d];
    // lane holds column j = jt*16+q, rows v + 8*hi of the subtile.
#define EC_EPILOGUE(MT, C0, C1, C2)                                            \
    {                                                                          \
      float ax[8] = {}, ay[8] = {}, az[8] = {};                                \
      _Pragma("unroll")                                                        \
      for (int jt = 0; jt < 3; ++jt) {                                         \
        v8f cc = (jt == 0) ? (C0) : ((jt == 1) ? (C1) : (C2));                 \
        int j = jt * 16 + q;                                                   \
        float bias = b2[i * 48 + j];                                           \
        _Pragma("unroll")                                                      \
        for (int v = 0; v < 8; ++v) {                                          \
          int el = (MT) * 16 + v + 8 * hi;                                     \
          float rw = cc[v] + bias;                                             \
          const float* t = sTMP + el * TMP_STRIDE + j * D2R;                   \
          ax[v] += rw * t[0];                                                  \
          ay[v] += rw * t[1];                                                  \
          az[v] += rw * t[2];                                                  \
        }                                                                      \
      }                                                                        \
      _Pragma("unroll")                                                        \
      for (int m = 1; m < 16; m <<= 1) {                                       \
        _Pragma("unroll")                                                      \
        for (int v = 0; v < 8; ++v) {                                          \
          ax[v] += __shfl_xor(ax[v], m, 16);                                   \
          ay[v] += __shfl_xor(ay[v], m, 16);                                   \
          az[v] += __shfl_xor(az[v], m, 16);                                   \
        }                                                                      \
      }                                                                        \
      if (q == 0) {                                                            \
        _Pragma("unroll")                                                      \
        for (int v = 0; v < 8; ++v) {                                          \
          int ge = e0 + (MT) * 16 + v + 8 * hi;                                \
          if (ge < E) {                                                        \
            size_t o = ((size_t)ge * M2R + i) * D2R;                           \
            out[o + 0] = ax[v];                                                \
            out[o + 1] = ay[v];                                                \
            out[o + 2] = az[v];                                                \
          }                                                                    \
        }                                                                      \
      }                                                                        \
    }

    EC_EPILOGUE(0, c00, c01, c02)
    EC_EPILOGUE(1, c10, c11, c12)
    EC_EPILOGUE(2, c20, c21, c22)
    EC_EPILOGUE(3, c30, c31, c32)
#undef EC_EPILOGUE
  }
}

extern "C" void kernel_launch(void* const* d_in, const int* in_sizes, int n_in,
                              void* d_out, int out_size, void* d_ws, size_t ws_size,
                              hipStream_t stream) {
  (void)n_in; (void)out_size; (void)ws_size;
  const int*   U     = (const int*)  d_in[0];
  const float* basis = (const float*)d_in[1];
  const float* ef    = (const float*)d_in[2];
  const float* f     = (const float*)d_in[3];
  const float* W1    = (const float*)d_in[4];
  const float* b1    = (const float*)d_in[5];
  const float* W2    = (const float*)d_in[6];
  const float* b2    = (const float*)d_in[7];
  float* out = (float*)d_out;
  const int E = in_sizes[0];

  _Float16* W1t = (_Float16*)d_ws;              //  128*32 halves =   8 KB
  _Float16* W2t = W1t + HIDDEN * EDGE_DIM;      //  768*128 halves = 192 KB

  ec_prep_kernel<<<400, 256, 0, stream>>>(W1, W2, W1t, W2t);

  int nblocks = (E + E_TILE - 1) / E_TILE;
  ec_main_kernel<<<nblocks, 256, 0, stream>>>(U, basis, ef, f, b1, b2,
                                              W1t, W2t, out, E);
}
